// TSSA_21148418965532
// MI455X (gfx1250) — matrix operand order repaired and checked by reading, not compiled, not run on hardware
//
#include <hip/hip_runtime.h>
#include <hip/hip_bf16.h>

// ---------------------------------------------------------------------------
// TSSA attention on MI455X (gfx1250).
//  GEMM1:  Wt[b,c,n] = sum_k qkv_w[c,k] * x[b,k,n]        (bf16 WMMA, f32 acc)
//  stats:  rn2, per-token head stat, softmax over heads, dots/attn  (fp32 VALU)
//  GEMM2:  out[b,o,n] = sum_c out_w[o,c] * ( -Wt[b,c,n]*Pi[b,h,n]*attn[b,c] )
//          + out_b[o]   -- modulation FUSED into the B-tile loader, saving a
//          full 100MB read + 100MB write of the intermediate through HBM.
// Both GEMMs share one kernel: D[o,n] = Wmat[o,:] . S[:,n], per-batch S/D.
// ---------------------------------------------------------------------------

#define BATCH 32
#define CDIM  768
#define NTOK  1024
#define NHEAD 12
#define DHEAD 64

#define TILE_M 64
#define TILE_N 64
#define TILE_K 32

typedef __attribute__((ext_vector_type(16))) __bf16        v16bf;
typedef __attribute__((ext_vector_type(8)))  float         v8f;
typedef __attribute__((ext_vector_type(8)))  unsigned int  v8u;

__device__ __forceinline__ unsigned short f2bf(float f) {
  unsigned u = __builtin_bit_cast(unsigned, f);
  u += 0x7FFFu + ((u >> 16) & 1u);           // round-to-nearest-even
  return (unsigned short)(u >> 16);
}

// ---------------------------------------------------------------------------
// Tiled bf16 WMMA GEMM: D[z][m][n] = sum_k Wmat[m][k]*Sval(z,k,n) (+bias[m]).
// MOD=true applies Sval = -S * Pi[z, k/64, n] * attn[z, k] in the loader.
// Block = 128 threads = 4 waves in a 2x2 grid; each wave owns 32x32 of a
// 64x64 output tile via 2x2 v_wmma_f32_16x16x32_bf16 with persistent v8f acc.
// ---------------------------------------------------------------------------
template <bool MOD>
__global__ __launch_bounds__(128) void gemm_bf16_wmma(
    const float* __restrict__ Wmat,   // [M][K] row-major
    const float* __restrict__ S,      // [Z][K][N] row-major
    float* __restrict__ D,            // [Z][M][N] row-major
    const float* __restrict__ bias,   // [M] or nullptr
    const float* __restrict__ Piv,    // [Z][NHEAD][N]  (MOD only)
    const float* __restrict__ attnv,  // [Z][K]         (MOD only)
    int M, int N, int K)
{
  __shared__ __align__(16) unsigned short Atile[TILE_M][TILE_K + 2]; // [m][k]
  __shared__ __align__(16) unsigned short Btile[TILE_N][TILE_K + 2]; // [n][k] (transposed)

  const int t     = threadIdx.x;
  const int lane  = t & 31;
  const int wave  = t >> 5;
  const int wm    = wave >> 1;        // 0..1
  const int wn    = wave & 1;         // 0..1
  const int halfl = lane >> 4;        // 0..1
  const int l16   = lane & 15;

  const int m0 = blockIdx.y * TILE_M;
  const int n0 = blockIdx.x * TILE_N;
  const size_t sb = (size_t)blockIdx.z * (size_t)K * (size_t)N;
  const size_t db = (size_t)blockIdx.z * (size_t)M * (size_t)N;

  v8f acc[2][2] = {};

  for (int k0 = 0; k0 < K; k0 += TILE_K) {
    // ---- stage A 64x32 (weights, fp32 -> bf16) ----
#pragma unroll
    for (int i = 0; i < 16; ++i) {
      int e = i * 128 + t;
      int m = e >> 5, k = e & 31;
      Atile[m][k] = f2bf(Wmat[(size_t)(m0 + m) * K + (k0 + k)]);
    }
    // ---- stage B 32x64 (activations), transposed into LDS [n][k] ----
#pragma unroll
    for (int i = 0; i < 16; ++i) {
      int e = i * 128 + t;
      int k = e >> 6, n = e & 63;
      float v = S[sb + (size_t)(k0 + k) * N + (n0 + n)];
      if (MOD) {
        int c = k0 + k;
        v *= -attnv[(size_t)blockIdx.z * CDIM + c] *
              Piv[((size_t)blockIdx.z * NHEAD + (c >> 6)) * NTOK + (n0 + n)];
      }
      Btile[n][k] = f2bf(v);
    }
    // prefetch next K-tiles into GL2 (gfx1250 global_prefetch_b8)
    if (k0 + TILE_K < K && t < 2) {
      __builtin_prefetch(&Wmat[(size_t)m0 * K + (k0 + TILE_K)], 0, 1);
      __builtin_prefetch(&S[sb + (size_t)(k0 + TILE_K) * N + n0], 0, 1);
    }
    __syncthreads();

    // ---- build fragments (ISA 7.12.2 16-bit A 16x32 / B 32x16 layouts) ----
    v16bf afrag[2], bfrag[2];
#pragma unroll
    for (int tm = 0; tm < 2; ++tm) {
      int m = wm * 32 + tm * 16 + l16;
      v8u r;
#pragma unroll
      for (int p = 0; p < 8; ++p) {
        int kk = ((p & 4) << 2) + halfl * 8 + ((2 * p) & 7);
        r[p] = *(const unsigned int*)&Atile[m][kk];
      }
      afrag[tm] = __builtin_bit_cast(v16bf, r);
    }
#pragma unroll
    for (int tn = 0; tn < 2; ++tn) {
      int n = wn * 32 + tn * 16 + l16;
      v8u r;
#pragma unroll
      for (int p = 0; p < 8; ++p) {
        int kk = ((p & 4) << 2) + halfl * 8 + ((2 * p) & 7);
        r[p] = *(const unsigned int*)&Btile[n][kk];
      }
      bfrag[tn] = __builtin_bit_cast(v16bf, r);
    }

#pragma unroll
    for (int tm = 0; tm < 2; ++tm)
#pragma unroll
      for (int tn = 0; tn < 2; ++tn)
        acc[tm][tn] = __builtin_amdgcn_wmma_f32_16x16x32_bf16(
            false, afrag[tm], false, bfrag[tn],
            (short)0, acc[tm][tn], false, false);
    __syncthreads();
  }

  // ---- epilogue: C/D f32 layout (VGPR r -> row r + 8*half) ----
#pragma unroll
  for (int tm = 0; tm < 2; ++tm) {
#pragma unroll
    for (int tn = 0; tn < 2; ++tn) {
#pragma unroll
      for (int r = 0; r < 8; ++r) {
        int m = m0 + wm * 32 + tm * 16 + r + 8 * halfl;
        int n = n0 + wn * 32 + tn * 16 + l16;
        float v = acc[tm][tn][r];
        if (bias) v += bias[m];
        D[db + (size_t)m * N + n] = v;
      }
    }
  }
}

// ---------------------------------------------------------------------------
// rn2[b*C+c] = 1 / max( sum_n Wt[b,c,n]^2 , 1e-24 )   (== 1/max(norm,1e-12)^2)
// ---------------------------------------------------------------------------
__global__ __launch_bounds__(256) void rowssq_kernel(
    const float* __restrict__ Wt, float* __restrict__ rn2)
{
  __shared__ float red[256];
  const int t = threadIdx.x;
  const float* p = Wt + (size_t)blockIdx.x * NTOK;
  float a = 0.f;
#pragma unroll
  for (int i = t; i < NTOK; i += 256) { float v = p[i]; a += v * v; }
  red[t] = a; __syncthreads();
  for (int s = 128; s > 0; s >>= 1) { if (t < s) red[t] += red[t + s]; __syncthreads(); }
  if (t == 0) rn2[blockIdx.x] = 1.0f / fmaxf(red[0], 1e-24f);
}

// ---------------------------------------------------------------------------
// Per (b,n): s[h] = sum_d Wt^2 * rn2 ; Pi[b,h,n] = softmax_h(s[h]*temp[h]).
// One thread per token; loads are coalesced across n.
// ---------------------------------------------------------------------------
__global__ __launch_bounds__(256) void stat_softmax_kernel(
    const float* __restrict__ Wt, const float* __restrict__ rn2,
    const float* __restrict__ temp, float* __restrict__ Pi)
{
  const int idx = blockIdx.x * 256 + threadIdx.x;   // 0 .. B*NTOK-1
  const int b = idx >> 10;
  const int n = idx & (NTOK - 1);
  const float* wb = Wt + (size_t)b * CDIM * NTOK + n;
  const float* r2 = rn2 + b * CDIM;

  float s[NHEAD];
#pragma unroll
  for (int h = 0; h < NHEAD; ++h) {
    float acc = 0.f;
#pragma unroll 4
    for (int d = 0; d < DHEAD; ++d) {
      int c = h * DHEAD + d;
      float w = wb[(size_t)c * NTOK];
      acc += w * w * r2[c];
    }
    s[h] = acc * temp[h];
  }
  float mx = s[0];
#pragma unroll
  for (int h = 1; h < NHEAD; ++h) mx = fmaxf(mx, s[h]);
  float e[NHEAD]; float sum = 0.f;
#pragma unroll
  for (int h = 0; h < NHEAD; ++h) { e[h] = __expf(s[h] - mx); sum += e[h]; }
  float inv = 1.0f / sum;
#pragma unroll
  for (int h = 0; h < NHEAD; ++h)
    Pi[((size_t)b * NHEAD + h) * NTOK + n] = e[h] * inv;
}

// ---------------------------------------------------------------------------
// Per (b,h): sumPi, dots[d] = (sum_n Pi*w^2)/(sumPi+1e-8), attn = 1/(1+dots).
// ---------------------------------------------------------------------------
__global__ __launch_bounds__(256) void dots_attn_kernel(
    const float* __restrict__ Wt, const float* __restrict__ Pi,
    float* __restrict__ attn)
{
  __shared__ float sPi[NTOK];
  __shared__ float red[256];
  const int bh = blockIdx.x;
  const int b = bh / NHEAD, h = bh % NHEAD;
  const int t = threadIdx.x;
  const float* prow  = Pi + (size_t)bh * NTOK;
  const float* wbase = Wt + ((size_t)b * CDIM + h * DHEAD) * NTOK;

  float psum = 0.f;
  for (int i = t; i < NTOK; i += 256) { float p = prow[i]; sPi[i] = p; psum += p; }
  red[t] = psum; __syncthreads();
  for (int s = 128; s > 0; s >>= 1) { if (t < s) red[t] += red[t + s]; __syncthreads(); }
  const float inv = 1.0f / (red[0] + 1e-8f);
  __syncthreads();

  for (int d = 0; d < DHEAD; ++d) {
    const float* wr = wbase + (size_t)d * NTOK;
    float pp = 0.f;
    for (int i = t; i < NTOK; i += 256) { float w = wr[i]; pp += sPi[i] * w * w; }
    red[t] = pp; __syncthreads();
    for (int s = 128; s > 0; s >>= 1) { if (t < s) red[t] += red[t + s]; __syncthreads(); }
    if (t == 0) {
      float dots = red[0] * inv;
      attn[(size_t)b * CDIM + h * DHEAD + d] = 1.0f / (1.0f + dots);
    }
    __syncthreads();
  }
}

// ---------------------------------------------------------------------------
extern "C" void kernel_launch(void* const* d_in, const int* in_sizes, int n_in,
                              void* d_out, int out_size, void* d_ws, size_t ws_size,
                              hipStream_t stream) {
  (void)in_sizes; (void)n_in; (void)out_size; (void)ws_size;
  const float* x     = (const float*)d_in[0];  // [B, C, H, W] = [B, C, N]
  const float* qkv_w = (const float*)d_in[1];  // [C, C]
  const float* temp  = (const float*)d_in[2];  // [HEADS]
  const float* out_w = (const float*)d_in[3];  // [C, C]
  const float* out_b = (const float*)d_in[4];  // [C]
  float* out = (float*)d_out;                  // [B, C, N]

  float* Wt   = (float*)d_ws;                              // B*C*N
  float* Pi   = Wt  + (size_t)BATCH * CDIM * NTOK;         // B*NHEAD*N
  float* rn2  = Pi  + (size_t)BATCH * NHEAD * NTOK;        // B*C
  float* attn = rn2 + (size_t)BATCH * CDIM;                // B*C

  dim3 ggrid(NTOK / TILE_N, CDIM / TILE_M, BATCH);         // 16 x 12 x 32

  // GEMM1: Wt[b,c,n] = qkv_w . x_b
  gemm_bf16_wmma<false><<<ggrid, 128, 0, stream>>>(
      qkv_w, x, Wt, nullptr, nullptr, nullptr, CDIM, NTOK, CDIM);
  // middle stage (fp32)
  rowssq_kernel<<<BATCH * CDIM, 256, 0, stream>>>(Wt, rn2);
  stat_softmax_kernel<<<(BATCH * NTOK) / 256, 256, 0, stream>>>(Wt, rn2, temp, Pi);
  dots_attn_kernel<<<BATCH * NHEAD, 256, 0, stream>>>(Wt, Pi, attn);
  // GEMM2 with fused modulation: out = out_w . (-Wt*Pi*attn) + out_b
  gemm_bf16_wmma<true><<<ggrid, 128, 0, stream>>>(
      out_w, Wt, out, out_b, Pi, attn, CDIM, NTOK, CDIM);
}
